// MultiHeadDotProductAttention_77309411545
// MI455X (gfx1250) — matrix operand order repaired and checked by reading, not compiled, not run on hardware
//
#include <hip/hip_runtime.h>
#include <hip/hip_bf16.h>

// ---- problem constants ----
#define BATCH 2
#define SEQ   2048
#define DMODEL 1024
#define HEADS 16
#define HDIM  64
#define NQKV  (HEADS*HDIM)      // 1024
#define MROWS (BATCH*SEQ)       // 4096

typedef __bf16 bf16_t;
typedef __attribute__((ext_vector_type(4)))  __bf16 bf16x4;
typedef __attribute__((ext_vector_type(8)))  __bf16 bf16x8;
typedef __attribute__((ext_vector_type(16))) __bf16 bf16x16;
typedef __attribute__((ext_vector_type(8)))  float  f32x8;

// ---------- WMMA helpers ----------
__device__ __forceinline__ f32x8 wmma_bf16(bf16x16 a, bf16x16 b, f32x8 c) {
    return __builtin_amdgcn_wmma_f32_16x16x32_bf16(
        /*neg_a=*/false, a, /*neg_b=*/false, b,
        /*c_mod=*/(short)0, c, /*reuse_a=*/false, /*reuse_b=*/false);
}

// A fragment (16x32 bf16, MxK): lane<16 holds row=lane, K 0..15;
// lane>=16 holds row=lane-16, K 16..31. Caller passes per-lane base
// p = row_ptr + k0 + ((lane>>4)<<4); second 8 elements live at +8 elems.
__device__ __forceinline__ bf16x16 load_a_frag(const bf16_t* p) {
    bf16x8 lo = *(const bf16x8*)(p);
    bf16x8 hi = *(const bf16x8*)(p + 8);
    return __builtin_shufflevector(lo, hi, 0,1,2,3,4,5,6,7,8,9,10,11,12,13,14,15);
}

// B fragment (32x16 bf16, KxN): lane<16 holds col=lane, K 0..15;
// lane>=16 holds col=lane-16, K 16..31 — 16 contiguous elements per lane.
__device__ __forceinline__ bf16x16 load_b_frag16(const bf16_t* p) {
    bf16x8 lo = *(const bf16x8*)(p);
    bf16x8 hi = *(const bf16x8*)(p + 8);
    return __builtin_shufflevector(lo, hi, 0,1,2,3,4,5,6,7,8,9,10,11,12,13,14,15);
}

// half-wave (16-lane) reductions; C-matrix rows live in lanes 0-15 / 16-31
__device__ __forceinline__ float hred_max(float v) {
    #pragma unroll
    for (int m = 1; m < 16; m <<= 1) v = fmaxf(v, __shfl_xor(v, m, 32));
    return v;
}
__device__ __forceinline__ float hred_sum(float v) {
    #pragma unroll
    for (int m = 1; m < 16; m <<= 1) v += __shfl_xor(v, m, 32);
    return v;
}

// ---------- prep kernels ----------
__global__ void cvt_f32_bf16_kernel(const float* __restrict__ src,
                                    bf16_t* __restrict__ dst, int n4) {
    int i = blockIdx.x * blockDim.x + threadIdx.x;
    if (i >= n4) return;
    const float4 f = ((const float4*)src)[i];
    bf16x4 o = { (bf16_t)f.x, (bf16_t)f.y, (bf16_t)f.z, (bf16_t)f.w };
    ((bf16x4*)dst)[i] = o;
}

// Pack row-major f32 W[K][N] into bf16 B-fragment tile order:
// tile = nt*(K/32)+kt holds 512 elems:
//   elem[lane*16+i] = W[kt*32 + (lane>>4)*16 + i][nt*16 + (lane&15)]
__global__ void pack_b_kernel(const float* __restrict__ src,
                              bf16_t* __restrict__ dst, int K, int N) {
    int e = blockIdx.x * blockDim.x + threadIdx.x;
    if (e >= K * N) return;
    const int tile = e >> 9;
    const int w    = e & 511;
    const int lane = w >> 4;
    const int i    = w & 15;
    const int ktiles = K >> 5;
    const int kt = tile % ktiles;
    const int nt = tile / ktiles;
    const int k = kt * 32 + ((lane >> 4) << 4) + i;
    const int n = nt * 16 + (lane & 15);
    dst[e] = (bf16_t)src[(size_t)k * N + n];
}

// ---------- fused QKV projection ----------
// A = xb [4096 x 1024] bf16 row-major; B = packed wq|wk|wv (B-fragment order).
// Each wave computes a 32(M) x 64(N) tile, software-pipelined one K-step deep.
// The 8 waves of a block share one N-strip (identical B stream -> WGP$ hits).
// Out: Q,K -> [B,H,S,HD] bf16 (Q pre-scaled by 1/8), V -> [B,H,HD,S] bf16.
__global__ void __launch_bounds__(256)
gemm_qkv_kernel(const bf16_t* __restrict__ xb, const bf16_t* __restrict__ wpk,
                const float* __restrict__ bq, const float* __restrict__ bk,
                const float* __restrict__ bv,
                bf16_t* __restrict__ qb, bf16_t* __restrict__ kb,
                bf16_t* __restrict__ vt) {
    const int lane = threadIdx.x & 31;
    const int gw = blockIdx.x * 8 + (threadIdx.x >> 5);   // 0..6143
    const int mt     = gw & 127;         // fastest: block = 8 consecutive m-tiles
    const int nstrip = (gw >> 7) & 15;
    const int mat    = gw >> 11;         // 0=Q 1=K 2=V
    const int hi = lane >> 4, ln = lane & 15;
    const int m0 = mt * 32;

    const bf16_t* wp  = wpk + (size_t)mat * DMODEL * NQKV;
    const bf16_t* ap0 = xb + (size_t)(m0 + ln) * DMODEL + (hi << 4);
    const bf16_t* ap1 = ap0 + (size_t)16 * DMODEL;
    const bf16_t* bp0 = wp + ((size_t)(nstrip * 4) * 32) * 512 + lane * 16;

    // prime the pipeline (kt = 0)
    bf16x16 a0c = load_a_frag(ap0);
    bf16x16 a1c = load_a_frag(ap1);
    bf16x16 bc[4];
    #pragma unroll
    for (int j = 0; j < 4; ++j) bc[j] = load_b_frag16(bp0 + (size_t)j * 32 * 512);

    f32x8 acc[2][4] = {};
    #pragma unroll 2
    for (int kt = 0; kt < 32; ++kt) {
        const int ktn = (kt < 31) ? kt + 1 : 31;
        const bf16_t* bpn = bp0 + (size_t)ktn * 512;
        __builtin_prefetch((const void*)(ap0 + (kt + 8) * 32), 0, 1);
        __builtin_prefetch((const void*)(bpn + 8 * 512), 0, 1);
        // issue next iteration's loads before consuming current fragments
        bf16x16 a0n = load_a_frag(ap0 + ktn * 32);
        bf16x16 a1n = load_a_frag(ap1 + ktn * 32);
        bf16x16 bn0 = load_b_frag16(bpn);
        bf16x16 bn1 = load_b_frag16(bpn + 32 * 512);
        bf16x16 bn2 = load_b_frag16(bpn + 64 * 512);
        bf16x16 bn3 = load_b_frag16(bpn + 96 * 512);

        acc[0][0] = wmma_bf16(a0c, bc[0], acc[0][0]);
        acc[1][0] = wmma_bf16(a1c, bc[0], acc[1][0]);
        acc[0][1] = wmma_bf16(a0c, bc[1], acc[0][1]);
        acc[1][1] = wmma_bf16(a1c, bc[1], acc[1][1]);
        acc[0][2] = wmma_bf16(a0c, bc[2], acc[0][2]);
        acc[1][2] = wmma_bf16(a1c, bc[2], acc[1][2]);
        acc[0][3] = wmma_bf16(a0c, bc[3], acc[0][3]);
        acc[1][3] = wmma_bf16(a1c, bc[3], acc[1][3]);

        a0c = a0n; a1c = a1n;
        bc[0] = bn0; bc[1] = bn1; bc[2] = bn2; bc[3] = bn3;
    }

    const float* bias = (mat == 0) ? bq : (mat == 1) ? bk : bv;
    const float qs = (mat == 0) ? 0.125f : 1.0f;   // 1/sqrt(64) folded into Q
    #pragma unroll
    for (int j = 0; j < 4; ++j) {
        const int n = nstrip * 64 + j * 16 + ln;
        const int h = n >> 6, hd = n & 63;
        const float bn = bias[n];
        #pragma unroll
        for (int half = 0; half < 2; ++half) {
            #pragma unroll
            for (int r = 0; r < 8; ++r) {
                const int m = m0 + half * 16 + r + (hi << 3);
                const int bb = m >> 11, ss = m & 2047;
                const float v = (acc[half][j][r] + bn) * qs;
                if (mat == 0)
                    qb[(((size_t)(bb * HEADS + h)) * SEQ + ss) * HDIM + hd] = (bf16_t)v;
                else if (mat == 1)
                    kb[(((size_t)(bb * HEADS + h)) * SEQ + ss) * HDIM + hd] = (bf16_t)v;
                else
                    vt[(((size_t)(bb * HEADS + h)) * HDIM + hd) * SEQ + ss] = (bf16_t)v;
            }
        }
    }
}

// ---------- flash attention: one wave per (b,h,16-query tile) ----------
// Pipelined: K fragments for block j0+32 and V fragments for j0 are issued
// before the softmax VALU section, which hides their latency.
__global__ void __launch_bounds__(256)
attn_kernel(const bf16_t* __restrict__ qb, const bf16_t* __restrict__ kb,
            const bf16_t* __restrict__ vt, bf16_t* __restrict__ ao) {
    __shared__ __align__(16) bf16_t pbuf[8][16 * 32];
    const int lane = threadIdx.x & 31;
    const int wave = threadIdx.x >> 5;
    const int gw = blockIdx.x * 8 + wave;        // 0..4095
    const int qt = gw & 127;
    const int h  = (gw >> 7) & 15;
    const int b  = gw >> 11;
    const int hi = lane >> 4, ln = lane & 15;
    const int q0 = qt * 16;

    const bf16_t* qh = qb + ((size_t)(b * HEADS + h)) * SEQ * HDIM;
    const bf16_t* kh = kb + ((size_t)(b * HEADS + h)) * SEQ * HDIM;
    const bf16_t* vh = vt + ((size_t)(b * HEADS + h)) * HDIM * SEQ;
    bf16_t* pw = &pbuf[wave][0];

    // Q fragments for the whole tile (HD=64 -> two 16x32 A-frags), Q pre-scaled
    const bf16x16 qa0 = load_a_frag(qh + (size_t)(q0 + ln) * HDIM + (hi << 4));
    const bf16x16 qa1 = load_a_frag(qh + (size_t)(q0 + ln) * HDIM + 32 + (hi << 4));

    f32x8 o0 = {}, o1 = {}, o2 = {}, o3 = {};
    float mrun[8], lrun[8];
    #pragma unroll
    for (int r = 0; r < 8; ++r) { mrun[r] = -3.0e38f; lrun[r] = 0.0f; }

    // prime K fragments for j0 = 0
    bf16x16 kf0 = load_b_frag16(kh + (size_t)(0 + ln) * HDIM + (hi << 4));
    bf16x16 kf1 = load_b_frag16(kh + (size_t)(0 + ln) * HDIM + 32 + (hi << 4));
    bf16x16 kf2 = load_b_frag16(kh + (size_t)(16 + ln) * HDIM + (hi << 4));
    bf16x16 kf3 = load_b_frag16(kh + (size_t)(16 + ln) * HDIM + 32 + (hi << 4));

    for (int j0 = 0; j0 < SEQ; j0 += 32) {
        // ---- scores: 16 queries x 32 keys (fragments loaded last iteration) ----
        f32x8 s0 = {}, s1 = {};
        s0 = wmma_bf16(qa0, kf0, s0);
        s0 = wmma_bf16(qa1, kf1, s0);
        s1 = wmma_bf16(qa0, kf2, s1);
        s1 = wmma_bf16(qa1, kf3, s1);

        // ---- issue next K block + this block's V before the softmax VALU ----
        const int jn = (j0 + 32 < SEQ) ? j0 + 32 : j0;
        const bf16_t* kn0p = kh + (size_t)(jn + ln) * HDIM + (hi << 4);
        const bf16_t* kn1p = kh + (size_t)(jn + 16 + ln) * HDIM + (hi << 4);
        bf16x16 kn0 = load_b_frag16(kn0p);
        bf16x16 kn1 = load_b_frag16(kn0p + 32);
        bf16x16 kn2 = load_b_frag16(kn1p);
        bf16x16 kn3 = load_b_frag16(kn1p + 32);
        const bf16_t* vp = vh + (size_t)ln * SEQ + j0 + (hi << 4);
        bf16x16 vf0 = load_b_frag16(vp);
        bf16x16 vf1 = load_b_frag16(vp + 16 * SEQ);
        bf16x16 vf2 = load_b_frag16(vp + 32 * SEQ);
        bf16x16 vf3 = load_b_frag16(vp + 48 * SEQ);

        // ---- online softmax update (overlaps outstanding loads) ----
        float p0[8], p1[8];
        #pragma unroll
        for (int r = 0; r < 8; ++r) {
            float tm = hred_max(fmaxf(s0[r], s1[r]));
            const float nm = fmaxf(mrun[r], tm);
            const float alpha = __expf(mrun[r] - nm);
            const float e0 = __expf(s0[r] - nm);
            const float e1 = __expf(s1[r] - nm);
            lrun[r] = lrun[r] * alpha + hred_sum(e0 + e1);
            mrun[r] = nm;
            p0[r] = e0; p1[r] = e1;
            o0[r] *= alpha; o1[r] *= alpha; o2[r] *= alpha; o3[r] *= alpha;
        }

        // ---- C-layout P -> LDS (row-major 16x32) -> A-layout fragment ----
        #pragma unroll
        for (int r = 0; r < 8; ++r) {
            const int m = r + (hi << 3);
            pw[m * 32 + ln]      = (bf16_t)p0[r];
            pw[m * 32 + 16 + ln] = (bf16_t)p1[r];
        }
        const bf16x16 pa = load_a_frag(pw + ln * 32 + (hi << 4));

        // ---- PV: 32 keys x 64 hd (V is [HD][S], contiguous per lane) ----
        o0 = wmma_bf16(pa, vf0, o0);
        o1 = wmma_bf16(pa, vf1, o1);
        o2 = wmma_bf16(pa, vf2, o2);
        o3 = wmma_bf16(pa, vf3, o3);

        kf0 = kn0; kf1 = kn1; kf2 = kn2; kf3 = kn3;
    }

    // ---- normalize + store to [B,S,H,HD] bf16 ----
    #pragma unroll
    for (int r = 0; r < 8; ++r) {
        const int m = r + (hi << 3);
        const float inv = 1.0f / lrun[r];
        bf16_t* dst = ao + (((size_t)(b * SEQ + q0 + m)) * HEADS + h) * HDIM + ln;
        dst[0]  = (bf16_t)(o0[r] * inv);
        dst[16] = (bf16_t)(o1[r] * inv);
        dst[32] = (bf16_t)(o2[r] * inv);
        dst[48] = (bf16_t)(o3[r] * inv);
    }
}

// ---------- output projection (32x64 tile per wave, pipelined) ----------
__global__ void __launch_bounds__(256)
gemm_out_kernel(const bf16_t* __restrict__ ab, const bf16_t* __restrict__ wpk,
                const float* __restrict__ bo, float* __restrict__ out) {
    const int lane = threadIdx.x & 31;
    const int gw = blockIdx.x * 8 + (threadIdx.x >> 5);   // 0..2047
    const int mt     = gw & 127;
    const int nstrip = gw >> 7;
    const int hi = lane >> 4, ln = lane & 15;
    const int m0 = mt * 32;

    const bf16_t* ap0 = ab + (size_t)(m0 + ln) * NQKV + (hi << 4);
    const bf16_t* ap1 = ap0 + (size_t)16 * NQKV;
    const bf16_t* bp0 = wpk + ((size_t)(nstrip * 4) * 32) * 512 + lane * 16;

    bf16x16 a0c = load_a_frag(ap0);
    bf16x16 a1c = load_a_frag(ap1);
    bf16x16 bc[4];
    #pragma unroll
    for (int j = 0; j < 4; ++j) bc[j] = load_b_frag16(bp0 + (size_t)j * 32 * 512);

    f32x8 acc[2][4] = {};
    #pragma unroll 2
    for (int kt = 0; kt < 32; ++kt) {
        const int ktn = (kt < 31) ? kt + 1 : 31;
        const bf16_t* bpn = bp0 + (size_t)ktn * 512;
        __builtin_prefetch((const void*)(ap0 + (kt + 8) * 32), 0, 1);
        __builtin_prefetch((const void*)(bpn + 8 * 512), 0, 1);
        bf16x16 a0n = load_a_frag(ap0 + ktn * 32);
        bf16x16 a1n = load_a_frag(ap1 + ktn * 32);
        bf16x16 bn0 = load_b_frag16(bpn);
        bf16x16 bn1 = load_b_frag16(bpn + 32 * 512);
        bf16x16 bn2 = load_b_frag16(bpn + 64 * 512);
        bf16x16 bn3 = load_b_frag16(bpn + 96 * 512);

        acc[0][0] = wmma_bf16(a0c, bc[0], acc[0][0]);
        acc[1][0] = wmma_bf16(a1c, bc[0], acc[1][0]);
        acc[0][1] = wmma_bf16(a0c, bc[1], acc[0][1]);
        acc[1][1] = wmma_bf16(a1c, bc[1], acc[1][1]);
        acc[0][2] = wmma_bf16(a0c, bc[2], acc[0][2]);
        acc[1][2] = wmma_bf16(a1c, bc[2], acc[1][2]);
        acc[0][3] = wmma_bf16(a0c, bc[3], acc[0][3]);
        acc[1][3] = wmma_bf16(a1c, bc[3], acc[1][3]);

        a0c = a0n; a1c = a1n;
        bc[0] = bn0; bc[1] = bn1; bc[2] = bn2; bc[3] = bn3;
    }

    #pragma unroll
    for (int j = 0; j < 4; ++j) {
        const int n = nstrip * 64 + j * 16 + ln;
        const float bn = bo[n];
        #pragma unroll
        for (int half = 0; half < 2; ++half) {
            #pragma unroll
            for (int r = 0; r < 8; ++r) {
                const int m = m0 + half * 16 + r + (hi << 3);
                out[(size_t)m * DMODEL + n] = acc[half][j][r] + bn;
            }
        }
    }
}

// ---------- host launch ----------
extern "C" void kernel_launch(void* const* d_in, const int* in_sizes, int n_in,
                              void* d_out, int out_size, void* d_ws, size_t ws_size,
                              hipStream_t stream) {
    const float* x  = (const float*)d_in[0];
    const float* wq = (const float*)d_in[1];
    const float* bq = (const float*)d_in[2];
    const float* wk = (const float*)d_in[3];
    const float* bk = (const float*)d_in[4];
    const float* wv = (const float*)d_in[5];
    const float* bv = (const float*)d_in[6];
    const float* wo = (const float*)d_in[7];
    const float* bo = (const float*)d_in[8];
    float* out = (float*)d_out;

    char* ws = (char*)d_ws;
    bf16_t* xb    = (bf16_t*)(ws);                         //  8 MiB
    bf16_t* wqkvp = (bf16_t*)(ws + (8u  << 20));           //  6 MiB (3 x 2 MiB)
    bf16_t* wop   = (bf16_t*)(ws + (14u << 20));           //  2 MiB
    bf16_t* qbuf  = (bf16_t*)(ws + (16u << 20));           //  8 MiB
    bf16_t* kbuf  = (bf16_t*)(ws + (24u << 20));           //  8 MiB
    bf16_t* vtb   = (bf16_t*)(ws + (32u << 20));           //  8 MiB
    bf16_t* aob   = (bf16_t*)(ws + (40u << 20));           //  8 MiB -> 48 MiB total

    // 1) convert x to bf16
    cvt_f32_bf16_kernel<<<(MROWS * DMODEL / 4 + 255) / 256, 256, 0, stream>>>(
        x, xb, MROWS * DMODEL / 4);

    // 2) pack weights into B-fragment order
    const int wElems = DMODEL * NQKV;
    const int packBlocks = (wElems + 255) / 256;
    pack_b_kernel<<<packBlocks, 256, 0, stream>>>(wq, wqkvp,              DMODEL, NQKV);
    pack_b_kernel<<<packBlocks, 256, 0, stream>>>(wk, wqkvp + wElems,     DMODEL, NQKV);
    pack_b_kernel<<<packBlocks, 256, 0, stream>>>(wv, wqkvp + 2 * wElems, DMODEL, NQKV);
    pack_b_kernel<<<packBlocks, 256, 0, stream>>>(wo, wop,                NQKV, DMODEL);

    // 3) fused QKV projection (3 mats * 128 m-tiles * 16 n-strips = 6144 waves)
    gemm_qkv_kernel<<<6144 / 8, 256, 0, stream>>>(xb, wqkvp, bq, bk, bv,
                                                  qbuf, kbuf, vtb);

    // 4) flash attention (2*16*128 = 4096 waves)
    attn_kernel<<<4096 / 8, 256, 0, stream>>>(qbuf, kbuf, vtb, aob);

    // 5) output projection (128 m-tiles * 16 n-strips = 2048 waves)
    gemm_out_kernel<<<2048 / 8, 256, 0, stream>>>(aob, wop, bo, out);
}
